// myAttention_26225070310145
// MI455X (gfx1250) — matrix operand order repaired and checked by reading, not compiled, not run on hardware
//
#include <hip/hip_runtime.h>
#include <hip/hip_bf16.h>

// ---------------------------------------------------------------------------
// EViT-style pruned attention for MI455X (gfx1250).
// bf16 WMMA pipeline + async global->LDS staging (ASYNCcnt) where available.
// ---------------------------------------------------------------------------

typedef unsigned short u16;
typedef unsigned int   u32;
typedef int v4i __attribute__((vector_size(16)));
typedef int v2i __attribute__((vector_size(8)));

typedef __attribute__((ext_vector_type(16))) __bf16 v16bf;
typedef __attribute__((ext_vector_type(8)))  float  v8f;

#define AS1 __attribute__((address_space(1)))
#define AS3 __attribute__((address_space(3)))

#if defined(__has_builtin)
#if __has_builtin(__builtin_amdgcn_global_load_async_to_lds_b128)
#define HAVE_ASYNC_LDS 1
#endif
#if __has_builtin(__builtin_amdgcn_global_load_async_to_lds_b64)
#define HAVE_ASYNC_LDS64 1
#endif
#endif
#ifndef HAVE_ASYNC_LDS
#define HAVE_ASYNC_LDS 0
#endif
#ifndef HAVE_ASYNC_LDS64
#define HAVE_ASYNC_LDS64 0
#endif

#define B_    64
#define N_    785
#define C_    768
#define H_    12
#define D_    64
#define KEEP  393
#define NSEL  392      // KEEP-1
#define NREM  784      // N-1
#define NPAD  800      // padded q/k/v rows (50 tiles of 16)
#define KVPAD 416      // padded selected keys (13 tiles of 32)
#define MROWS (B_*N_)  // 50240  (divisible by 32)

// ---- bf16 helpers ---------------------------------------------------------
__device__ __forceinline__ u16 f2bf(float f) {
  u32 u = __float_as_uint(f);
  u32 r = (u + 0x7FFFu + ((u >> 16) & 1u)) >> 16;
  return (u16)r;
}
__device__ __forceinline__ float bf2f(u16 h) {
  return __uint_as_float(((u32)h) << 16);
}

union Frag { v16bf v; u32 u[8]; };

__device__ __forceinline__ v8f wmma_bf16(const Frag& a, const Frag& b, v8f c) {
  return __builtin_amdgcn_wmma_f32_16x16x32_bf16(
      /*neg_a=*/false, a.v, /*neg_b=*/false, b.v,
      /*c_mod=*/(short)0, c, /*reuse_a=*/false, /*reuse_b=*/false);
}

__device__ __forceinline__ v8f zero8() {
  v8f z;
#pragma unroll
  for (int i = 0; i < 8; ++i) z[i] = 0.0f;
  return z;
}

// ---- async global -> LDS copies (ASYNCcnt path on gfx1250) ----------------
__device__ __forceinline__ void async_copy16(const u16* g, u16* l) {
#if HAVE_ASYNC_LDS
  __builtin_amdgcn_global_load_async_to_lds_b128(
      (AS1 v4i*)g, (AS3 v4i*)l, /*offset=*/0, /*cpol=*/0);
#else
  *(uint4*)l = *(const uint4*)g;
#endif
}
__device__ __forceinline__ void async_copy8(const u16* g, u16* l) {
#if HAVE_ASYNC_LDS64
  __builtin_amdgcn_global_load_async_to_lds_b64(
      (AS1 v2i*)g, (AS3 v2i*)l, /*offset=*/0, /*cpol=*/0);
#else
  *(uint2*)l = *(const uint2*)g;
#endif
}
__device__ __forceinline__ void wait_async_le0() {
#if HAVE_ASYNC_LDS || HAVE_ASYNC_LDS64
#if defined(__has_builtin) && __has_builtin(__builtin_amdgcn_s_wait_asynccnt)
  __builtin_amdgcn_s_wait_asynccnt(0);
#else
  asm volatile("s_wait_asynccnt 0" ::: "memory");
#endif
#endif
}
__device__ __forceinline__ void wait_async_le2() {
#if HAVE_ASYNC_LDS || HAVE_ASYNC_LDS64
#if defined(__has_builtin) && __has_builtin(__builtin_amdgcn_s_wait_asynccnt)
  __builtin_amdgcn_s_wait_asynccnt(2);
#else
  asm volatile("s_wait_asynccnt 2" ::: "memory");
#endif
#endif
}

// ---------------------------------------------------------------------------
// Kernel 0a: fp32 -> bf16 elementwise convert (x -> xb), vectorized x4.
// ---------------------------------------------------------------------------
__global__ __launch_bounds__(256) void evit_cvt_bf16(
    const float* __restrict__ in, u16* __restrict__ out, long n4) {
  long i = (long)blockIdx.x * 256 + threadIdx.x;
  if (i >= n4) return;
  float4 f = ((const float4*)in)[i];
  union { u16 h[4]; uint2 u; } p;
  p.h[0] = f2bf(f.x); p.h[1] = f2bf(f.y);
  p.h[2] = f2bf(f.z); p.h[3] = f2bf(f.w);
  ((uint2*)out)[i] = p.u;
}

// ---------------------------------------------------------------------------
// Kernel 0b: fp32 -> bf16 transpose (weights), W[rows][cols] -> WT[cols][rows]
// ---------------------------------------------------------------------------
__global__ __launch_bounds__(256) void evit_transpose_bf16(
    const float* __restrict__ W, u16* __restrict__ WT, int rows, int cols) {
  __shared__ __align__(16) float tile[32][33];
  int tx = threadIdx.x & 31, ty = threadIdx.x >> 5;
  int bx = blockIdx.x * 32;   // col base
  int by = blockIdx.y * 32;   // row base
#pragma unroll
  for (int i = 0; i < 32; i += 8) {
    int r = by + ty + i, c = bx + tx;
    tile[ty + i][tx] = (r < rows && c < cols) ? W[(size_t)r * cols + c] : 0.0f;
  }
  __syncthreads();
#pragma unroll
  for (int i = 0; i < 32; i += 8) {
    int r = bx + ty + i;  // output row = original col
    int c = by + tx;      // output col = original row
    if (r < cols && c < rows) WT[(size_t)r * rows + c] = f2bf(tile[tx][ty + i]);
  }
}

// ---------------------------------------------------------------------------
// Kernel 1: qkv = xb @ Wqkv  (M=50240, K=768, Nout=2304), WMMA bf16.
// Block 256 = 8 waves (2x4), tile 32(M) x 64(N). Double-buffered async LDS
// staging: every thread issues exactly one b64 (A) + one b128 (B) async op
// per K-tile, so the wait threshold is a uniform s_wait_asynccnt 2 (no
// divergent exec masking). Epilogue scatters into q/k/v [B][H][NPAD][64].
// ---------------------------------------------------------------------------
__global__ __launch_bounds__(256) void evit_gemm_qkv(
    const u16* __restrict__ A, const u16* __restrict__ WT,
    u16* __restrict__ qb, u16* __restrict__ kb, u16* __restrict__ vb) {
  __shared__ __align__(16) u16 sA[2][32][40];
  __shared__ __align__(16) u16 sB[2][64][40];

  const int tid = threadIdx.x;
  const int wave = tid >> 5, lane = tid & 31;
  const int wm = wave >> 2, wn = wave & 3;
  const int m0 = blockIdx.x * 32;
  const int n0 = blockIdx.y * 64;

  v8f acc = zero8();
  const int ar  = wm * 16 + (lane & 15);
  const int kh  = (lane >> 4) * 16;   // A K-half within 32
  const int bn  = wn * 16 + (lane & 15);
  const int bkh = (lane >> 4) * 16;   // B K-half within 32

  // staging coords: A in 8B chunks (32 rows x 8 chunks = 256 = all threads),
  //                 B in 16B chunks (64 rows x 4 chunks = 256 = all threads)
  const int arr = tid >> 3, acs = (tid & 7) * 4;   // A: row, col (u16 units)
  const int brr = tid >> 2, bcs = (tid & 3) * 8;   // B: row, col (u16 units)

  auto stageA = [&](int kt, int buf) {
    async_copy8(A + (size_t)(m0 + arr) * 768 + kt * 32 + acs, &sA[buf][arr][acs]);
  };
  auto stageB = [&](int kt, int buf) {
    async_copy16(WT + (size_t)(n0 + brr) * 768 + kt * 32 + bcs, &sB[buf][brr][bcs]);
  };

  stageA(0, 0); stageB(0, 0);
  for (int kt = 0; kt < 24; ++kt) {
    const int buf = kt & 1;
    const bool more = (kt + 1) < 24;
    if (more) {
      stageA(kt + 1, buf ^ 1); stageB(kt + 1, buf ^ 1);
      wait_async_le2();          // previous tile's group (2 ops/thread) done
    } else {
      wait_async_le0();
    }
    __syncthreads();

    Frag a, b;
#pragma unroll
    for (int j = 0; j < 4; ++j) {
      a.u[j]     = *(const u32*)&sA[buf][ar][kh + 2 * j];
      a.u[4 + j] = *(const u32*)&sA[buf][ar][8 + kh + 2 * j];
    }
#pragma unroll
    for (int j = 0; j < 8; ++j)
      b.u[j] = *(const u32*)&sB[buf][bn][bkh + 2 * j];
    acc = wmma_bf16(a, b, acc);
    __syncthreads();
  }

  // Epilogue: scatter to q/k/v  [B][H][NPAD][64]
  const int col = n0 + wn * 16 + (lane & 15);
  const int t   = col / 768;
  const int rst = col - t * 768;
  const int h   = rst >> 6;
  const int dd  = rst & 63;
  u16* dst = (t == 0) ? qb : (t == 1) ? kb : vb;
  const int rbase = m0 + wm * 16 + (lane >> 4) * 8;
#pragma unroll
  for (int r = 0; r < 8; ++r) {
    int row = rbase + r;
    int b_  = row / N_;
    int nn  = row - b_ * N_;
    dst[(((size_t)b_ * H_ + h) * NPAD + nn) * D_ + dd] = f2bf(acc[r]);
  }
}

// ---------------------------------------------------------------------------
// Kernel 2: per-row key norms  vals[b,h,r] = ||k[b,h,r+1,:]|| / 64
// ---------------------------------------------------------------------------
__global__ __launch_bounds__(256) void evit_knorms(
    const u16* __restrict__ kb, float* __restrict__ norms) {
  size_t i = (size_t)blockIdx.x * 256 + threadIdx.x;
  if (i >= (size_t)B_ * H_ * NREM) return;
  int bh = (int)(i / NREM);
  int r  = (int)(i - (size_t)bh * NREM);
  const u16* src = kb + ((size_t)bh * NPAD + (r + 1)) * D_;
  float s = 0.0f;
#pragma unroll 8
  for (int d = 0; d < D_; ++d) {
    float f = bf2f(src[d]);
    s += f * f;
  }
  norms[i] = sqrtf(s) * (1.0f / 64.0f);
}

// ---------------------------------------------------------------------------
// Kernel 3: per-(b,h) top-392 selection (bitonic), ascending index sort,
// gather new_k/new_v (row 0 = cls token, rows 393..415 zero padding).
// ---------------------------------------------------------------------------
__global__ __launch_bounds__(512) void evit_topk_gather(
    const float* __restrict__ norms,
    const u16* __restrict__ kb, const u16* __restrict__ vb,
    u16* __restrict__ nk, u16* __restrict__ nv) {
  __shared__ float sv[1024];
  __shared__ int   si[1024];
  __shared__ int   sidx[512];
  const int bh = blockIdx.x, tid = threadIdx.x;

  for (int i = tid; i < 1024; i += 512) {
    if (i < NREM) { sv[i] = norms[(size_t)bh * NREM + i]; si[i] = i; }
    else          { sv[i] = -3.0e38f; si[i] = 0x7fffffff; }
  }
  __syncthreads();

  // Bitonic sort, descending by value, tie -> smaller index first.
  for (int k = 2; k <= 1024; k <<= 1) {
    for (int j = k >> 1; j > 0; j >>= 1) {
      int ii = 2 * tid - (tid & (j - 1));
      int jj = ii + j;
      bool desc = ((ii & k) == 0);
      float va = sv[ii], vb2 = sv[jj];
      int   ia = si[ii], ib = si[jj];
      bool before = (va > vb2) || (va == vb2 && ia < ib);
      if (desc ? !before : before) {
        sv[ii] = vb2; sv[jj] = va; si[ii] = ib; si[jj] = ia;
      }
      __syncthreads();
    }
  }

  for (int i = tid; i < 512; i += 512)
    sidx[i] = (i < NSEL) ? si[i] : 0x7fffffff;
  __syncthreads();

  // Ascending bitonic sort of the 392 kept indices (padded to 512).
  for (int k = 2; k <= 512; k <<= 1) {
    for (int j = k >> 1; j > 0; j >>= 1) {
      if (tid < 256) {
        int ii = 2 * tid - (tid & (j - 1));
        int jj = ii + j;
        bool up = ((ii & k) == 0);
        int a = sidx[ii], b2 = sidx[jj];
        if (up ? (a > b2) : (a < b2)) { sidx[ii] = b2; sidx[jj] = a; }
      }
      __syncthreads();
    }
  }

  // Gather: row 0 = cls; rows 1..392 = kept; rows 393..415 = zeros.
  const u16* ksrc = kb + (size_t)bh * NPAD * D_;
  const u16* vsrc = vb + (size_t)bh * NPAD * D_;
  u16* kdst = nk + (size_t)bh * KVPAD * D_;
  u16* vdst = nv + (size_t)bh * KVPAD * D_;
  for (int e = tid; e < KVPAD * 32; e += 512) {   // 32 dwords per 64-elem row
    int r = e >> 5, c = e & 31;
    u32 vk = 0, vv = 0;
    int srcr = -1;
    if (r == 0) srcr = 0;
    else if (r <= NSEL) srcr = 1 + sidx[r - 1];
    if (srcr >= 0) {
      vk = *(const u32*)(ksrc + (size_t)srcr * D_ + c * 2);
      vv = *(const u32*)(vsrc + (size_t)srcr * D_ + c * 2);
    }
    *(u32*)(kdst + (size_t)r * D_ + c * 2) = vk;
    *(u32*)(vdst + (size_t)r * D_ + c * 2) = vv;
  }
}

// ---------------------------------------------------------------------------
// Kernel 4: flash attention over 393 pruned keys (13 tiles of 32, masked).
// Block = 128 threads (4 waves); each wave owns a 16-row Q tile. K tile is
// staged via async-to-LDS (2 uniform b128 ops/thread); V tile is manually
// transposed into LDS. Output: ao [B][N][C] bf16.
// ---------------------------------------------------------------------------
__global__ __launch_bounds__(128) void evit_attn(
    const u16* __restrict__ qb, const u16* __restrict__ nk,
    const u16* __restrict__ nv, u16* __restrict__ ao) {
  __shared__ __align__(16) u16 sK[32][72];      // [key][feat]
  __shared__ __align__(16) u16 sVt[64][36];     // [d][key] (transposed)
  __shared__ __align__(16) u16 sP[4][16][36];   // per-wave P tile (16x32)

  const int bh = blockIdx.x;
  const int b_ = bh / H_, h = bh % H_;
  const int tid = threadIdx.x, wave = tid >> 5, lane = tid & 31;
  const int qt = blockIdx.y * 4 + wave;
  const bool active = (qt < 50);
  const int qtc = active ? qt : 49;

  const int ar   = lane & 15;
  const int kh   = (lane >> 4) * 16;
  const int bn   = lane & 15;
  const int bkh  = (lane >> 4) * 16;
  const int prow = (lane >> 4) * 8;

  // Q fragments (d 0..31 and 32..63), held for whole kernel.
  const u16* qrow = qb + ((size_t)bh * NPAD + (size_t)qtc * 16) * D_;
  Frag qa0, qa1;
#pragma unroll
  for (int j = 0; j < 4; ++j) {
    qa0.u[j]     = *(const u32*)(qrow + (size_t)ar * 64 + kh + 2 * j);
    qa0.u[4 + j] = *(const u32*)(qrow + (size_t)ar * 64 + 8 + kh + 2 * j);
    qa1.u[j]     = *(const u32*)(qrow + (size_t)ar * 64 + 32 + kh + 2 * j);
    qa1.u[4 + j] = *(const u32*)(qrow + (size_t)ar * 64 + 40 + kh + 2 * j);
  }

  float m[8], l[8];
  v8f o0 = zero8(), o1 = zero8(), o2 = zero8(), o3 = zero8();
#pragma unroll
  for (int r = 0; r < 8; ++r) { m[r] = -3.0e38f; l[r] = 0.0f; }

  const u16* kbase = nk + (size_t)bh * KVPAD * D_;
  const u16* vbase = nv + (size_t)bh * KVPAD * D_;
  const float scale = 0.125f;   // 1/sqrt(64)

  for (int t = 0; t < KVPAD / 32; ++t) {
    __syncthreads();
    // Stage K tile (32 keys x 64 feats) via async-to-LDS, 16B chunks.
#pragma unroll
    for (int e = tid; e < 256; e += 128) {
      int r = e >> 3, c = (e & 7) * 8;
      async_copy16(kbase + ((size_t)(t * 32 + r)) * 64 + c, &sK[r][c]);
    }
    // Stage V tile transposed: sVt[d][key].
    for (int e = tid; e < 1024; e += 128) {
      int r = e >> 5, c2 = (e & 31) * 2;
      u32 pv = *(const u32*)(vbase + ((size_t)(t * 32 + r)) * 64 + c2);
      sVt[c2][r]     = (u16)(pv & 0xffff);
      sVt[c2 + 1][r] = (u16)(pv >> 16);
    }
    wait_async_le0();
    __syncthreads();

    // S = Q K^T  (16 x 32 via two 16x16 accumulators)
    v8f s0 = zero8(), s1 = zero8();
    Frag bk;
#pragma unroll
    for (int j = 0; j < 8; ++j) bk.u[j] = *(const u32*)&sK[bn][bkh + 2 * j];
    s0 = wmma_bf16(qa0, bk, s0);
#pragma unroll
    for (int j = 0; j < 8; ++j) bk.u[j] = *(const u32*)&sK[bn][32 + bkh + 2 * j];
    s0 = wmma_bf16(qa1, bk, s0);
#pragma unroll
    for (int j = 0; j < 8; ++j) bk.u[j] = *(const u32*)&sK[16 + bn][bkh + 2 * j];
    s1 = wmma_bf16(qa0, bk, s1);
#pragma unroll
    for (int j = 0; j < 8; ++j) bk.u[j] = *(const u32*)&sK[16 + bn][32 + bkh + 2 * j];
    s1 = wmma_bf16(qa1, bk, s1);

    // Online softmax (row stats via 16-lane shfl_xor reductions).
    const bool v0ok = (t * 32 + bn)      < KEEP;
    const bool v1ok = (t * 32 + 16 + bn) < KEEP;
    float alpha[8];
#pragma unroll
    for (int r = 0; r < 8; ++r) {
      float a  = v0ok ? s0[r] * scale : -3.0e38f;
      float b2 = v1ok ? s1[r] * scale : -3.0e38f;
      float mx = fmaxf(a, b2);
#pragma unroll
      for (int msk = 1; msk < 16; msk <<= 1)
        mx = fmaxf(mx, __shfl_xor(mx, msk, 32));
      float mnew = fmaxf(m[r], mx);
      alpha[r] = __expf(m[r] - mnew);
      m[r] = mnew;
      float p0 = v0ok ? __expf(a - mnew)  : 0.0f;
      float p1 = v1ok ? __expf(b2 - mnew) : 0.0f;
      float rs = p0 + p1;
#pragma unroll
      for (int msk = 1; msk < 16; msk <<= 1)
        rs += __shfl_xor(rs, msk, 32);
      l[r] = l[r] * alpha[r] + rs;
      sP[wave][prow + r][bn]      = f2bf(p0);
      sP[wave][prow + r][16 + bn] = f2bf(p1);
      o0[r] *= alpha[r]; o1[r] *= alpha[r];
      o2[r] *= alpha[r]; o3[r] *= alpha[r];
    }

    // O += P @ V  (P via LDS -> A-fragment layout; V B-frags per d-block)
    Frag pa, bv;
#pragma unroll
    for (int j = 0; j < 4; ++j) {
      pa.u[j]     = *(const u32*)&sP[wave][ar][kh + 2 * j];
      pa.u[4 + j] = *(const u32*)&sP[wave][ar][8 + kh + 2 * j];
    }
#pragma unroll
    for (int j = 0; j < 8; ++j) bv.u[j] = *(const u32*)&sVt[0 + bn][bkh + 2 * j];
    o0 = wmma_bf16(pa, bv, o0);
#pragma unroll
    for (int j = 0; j < 8; ++j) bv.u[j] = *(const u32*)&sVt[16 + bn][bkh + 2 * j];
    o1 = wmma_bf16(pa, bv, o1);
#pragma unroll
    for (int j = 0; j < 8; ++j) bv.u[j] = *(const u32*)&sVt[32 + bn][bkh + 2 * j];
    o2 = wmma_bf16(pa, bv, o2);
#pragma unroll
    for (int j = 0; j < 8; ++j) bv.u[j] = *(const u32*)&sVt[48 + bn][bkh + 2 * j];
    o3 = wmma_bf16(pa, bv, o3);
  }

  if (active) {
#pragma unroll
    for (int r = 0; r < 8; ++r) {
      int qrg = qt * 16 + prow + r;
      if (qrg < N_) {
        float inv = 1.0f / l[r];
        size_t base = ((size_t)b_ * N_ + qrg) * C_ + (size_t)h * 64;
        ao[base +  0 + bn] = f2bf(o0[r] * inv);
        ao[base + 16 + bn] = f2bf(o1[r] * inv);
        ao[base + 32 + bn] = f2bf(o2[r] * inv);
        ao[base + 48 + bn] = f2bf(o3[r] * inv);
      }
    }
  }
}

// ---------------------------------------------------------------------------
// Kernel 5: out = ao @ Wproj + bproj   (M=50240, K=768, N=768), fp32 out.
// Same uniform double-buffered async main loop as GEMM1.
// ---------------------------------------------------------------------------
__global__ __launch_bounds__(256) void evit_gemm_proj(
    const u16* __restrict__ A, const u16* __restrict__ WT,
    const float* __restrict__ bias, float* __restrict__ out) {
  __shared__ __align__(16) u16 sA[2][32][40];
  __shared__ __align__(16) u16 sB[2][64][40];

  const int tid = threadIdx.x;
  const int wave = tid >> 5, lane = tid & 31;
  const int wm = wave >> 2, wn = wave & 3;
  const int m0 = blockIdx.x * 32;
  const int n0 = blockIdx.y * 64;

  v8f acc = zero8();
  const int ar  = wm * 16 + (lane & 15);
  const int kh  = (lane >> 4) * 16;
  const int bn  = wn * 16 + (lane & 15);
  const int bkh = (lane >> 4) * 16;

  const int arr = tid >> 3, acs = (tid & 7) * 4;
  const int brr = tid >> 2, bcs = (tid & 3) * 8;

  auto stageA = [&](int kt, int buf) {
    async_copy8(A + (size_t)(m0 + arr) * 768 + kt * 32 + acs, &sA[buf][arr][acs]);
  };
  auto stageB = [&](int kt, int buf) {
    async_copy16(WT + (size_t)(n0 + brr) * 768 + kt * 32 + bcs, &sB[buf][brr][bcs]);
  };

  stageA(0, 0); stageB(0, 0);
  for (int kt = 0; kt < 24; ++kt) {
    const int buf = kt & 1;
    const bool more = (kt + 1) < 24;
    if (more) {
      stageA(kt + 1, buf ^ 1); stageB(kt + 1, buf ^ 1);
      wait_async_le2();
    } else {
      wait_async_le0();
    }
    __syncthreads();

    Frag a, b;
#pragma unroll
    for (int j = 0; j < 4; ++j) {
      a.u[j]     = *(const u32*)&sA[buf][ar][kh + 2 * j];
      a.u[4 + j] = *(const u32*)&sA[buf][ar][8 + kh + 2 * j];
    }
#pragma unroll
    for (int j = 0; j < 8; ++j)
      b.u[j] = *(const u32*)&sB[buf][bn][bkh + 2 * j];
    acc = wmma_bf16(a, b, acc);
    __syncthreads();
  }

  const int col = n0 + wn * 16 + (lane & 15);
  const float bv = bias[col];
  const int rbase = m0 + wm * 16 + (lane >> 4) * 8;
#pragma unroll
  for (int r = 0; r < 8; ++r)
    out[(size_t)(rbase + r) * 768 + col] = acc[r] + bv;
}

// ---------------------------------------------------------------------------
// Host launcher
// ---------------------------------------------------------------------------
extern "C" void kernel_launch(void* const* d_in, const int* in_sizes, int n_in,
                              void* d_out, int out_size, void* d_ws, size_t ws_size,
                              hipStream_t stream) {
  (void)in_sizes; (void)n_in; (void)out_size; (void)ws_size;
  const float* x     = (const float*)d_in[0];
  const float* Wqkv  = (const float*)d_in[1];
  const float* Wproj = (const float*)d_in[2];
  const float* bproj = (const float*)d_in[3];
  // d_in[4] = keep_len (393) — baked into KEEP.

  char* ws = (char*)d_ws;
  const size_t SZ_QKV   = (size_t)B_ * H_ * NPAD * D_ * sizeof(u16);   // 78,643,200
  const size_t SZ_WQKVT = (size_t)2304 * 768 * sizeof(u16);
  const size_t SZ_WPROJT= (size_t)768 * 768 * sizeof(u16);
  const size_t SZ_NORMS = (size_t)B_ * H_ * NREM * sizeof(float);
  const size_t SZ_KV    = (size_t)B_ * H_ * KVPAD * D_ * sizeof(u16);

  u16*   qb     = (u16*)(ws);
  u16*   kb     = (u16*)(ws + SZ_QKV);
  u16*   vb     = (u16*)(ws + 2 * SZ_QKV);
  u16*   wqkvT  = (u16*)(ws + 3 * SZ_QKV);
  u16*   wprojT = (u16*)(ws + 3 * SZ_QKV + SZ_WQKVT);
  float* norms  = (float*)(ws + 3 * SZ_QKV + SZ_WQKVT + SZ_WPROJT);
  u16*   nk     = (u16*)(ws + 3 * SZ_QKV + SZ_WQKVT + SZ_WPROJT + SZ_NORMS);
  u16*   nv     = (u16*)(ws + 3 * SZ_QKV + SZ_WQKVT + SZ_WPROJT + SZ_NORMS + SZ_KV);
  u16*   xb     = nk;   // x-as-bf16 lives in the (not yet written) nk/nv arena
  u16*   ao     = kb;   // kb is dead after the gather; reuse for attention output

  const long n4 = (long)MROWS * C_ / 4;
  evit_cvt_bf16<<<(int)((n4 + 255) / 256), 256, 0, stream>>>(x, xb, n4);

  evit_transpose_bf16<<<dim3(2304 / 32, 768 / 32), 256, 0, stream>>>(
      Wqkv, wqkvT, 768, 2304);
  evit_transpose_bf16<<<dim3(768 / 32, 768 / 32), 256, 0, stream>>>(
      Wproj, wprojT, 768, 768);

  evit_gemm_qkv<<<dim3(MROWS / 32, 2304 / 64), 256, 0, stream>>>(
      xb, wqkvT, qb, kb, vb);

  const int nnorm = B_ * H_ * NREM;
  evit_knorms<<<(nnorm + 255) / 256, 256, 0, stream>>>(kb, norms);

  evit_topk_gather<<<B_ * H_, 512, 0, stream>>>(norms, kb, vb, nk, nv);

  evit_attn<<<dim3(B_ * H_, 13), 128, 0, stream>>>(qb, nk, nv, ao);

  evit_gemm_proj<<<dim3(MROWS / 32, 768 / 64), 256, 0, stream>>>(
      ao, wprojT, bproj, (float*)d_out);
}
